// Net_32555852104136
// MI455X (gfx1250) — matrix operand order repaired and checked by reading, not compiled, not run on hardware
//
#include <hip/hip_runtime.h>
#include <cstdint>
#include <cstddef>

// ---------- CDNA5 WMMA types ----------
typedef __attribute__((ext_vector_type(16))) __bf16 v16bf;
typedef __attribute__((ext_vector_type(8)))  float  v8f;

#define NN 40000   // nodes
#define FF 200     // node features
#define EE 240000  // edges
#define BB 128     // graphs

// ---------- bf16 helpers (storage = ushort) ----------
__device__ __forceinline__ unsigned short f2bf(float f) {
    union { float f; unsigned int u; } v; v.f = f;
    unsigned int u = v.u;
    unsigned int r = u + 0x7FFFu + ((u >> 16) & 1u);   // RNE
    return (unsigned short)(r >> 16);
}
__device__ __forceinline__ float bf2f(unsigned short h) {
    union { unsigned int u; float f; } v; v.u = ((unsigned int)h) << 16;
    return v.f;
}

union ABfrag { v16bf v; uint4 q[2]; unsigned short u[16]; };

// ---------- CDNA5 async global->LDS copy (ASYNCcnt path) ----------
typedef int v4i_t __attribute__((vector_size(16)));
typedef __attribute__((address_space(1))) v4i_t* g_b128_ptr;
typedef __attribute__((address_space(3))) v4i_t* l_b128_ptr;
typedef __attribute__((address_space(1))) void*  g_void_ptr;
typedef __attribute__((address_space(3))) void*  l_void_ptr;

__device__ __forceinline__ void async_b128(const unsigned short* g, unsigned short* l) {
#if __has_builtin(__builtin_amdgcn_global_load_async_to_lds_b128)
    __builtin_amdgcn_global_load_async_to_lds_b128(
        (g_b128_ptr)(g_void_ptr)(void*)g,
        (l_b128_ptr)(l_void_ptr)(void*)l, 0, 0);
#else
    unsigned lo = (unsigned)(uintptr_t)(l_void_ptr)(void*)l;
    asm volatile("global_load_async_to_lds_b128 %0, %1, off"
                 :: "v"(lo), "v"(g) : "memory");
#endif
}
__device__ __forceinline__ void wait_async0() {
#if __has_builtin(__builtin_amdgcn_s_wait_asynccnt)
    __builtin_amdgcn_s_wait_asynccnt(0);
#else
    asm volatile("s_wait_asynccnt 0x0" ::: "memory");
#endif
}

// ---------- CSR offsets via binary search (index arrays are sorted) ----------
__global__ void lb_kernel(const int* __restrict__ arr, int len, int* __restrict__ off, int nseg) {
    int i = blockIdx.x * blockDim.x + threadIdx.x;
    if (i > nseg) return;
    if (i == nseg) { off[i] = len; return; }
    int lo = 0, hi = len;
    while (lo < hi) { int mid = (lo + hi) >> 1; if (arr[mid] < i) lo = mid + 1; else hi = mid; }
    off[i] = lo;
}

// ---------- pack fp32 weights (Ko x No) into WMMA B-fragment tile layout ----------
// tile = 32(K) x 16(N); tiles ordered kt-major then nt; within tile: lane*16+e,
// lane&15 = column, (lane>>4)*16 + e = k offset.
__global__ void pack_w(const float* __restrict__ W, int Ko, int No, int Kp, int Np,
                       unsigned short* __restrict__ dst) {
    size_t idx = (size_t)blockIdx.x * 256 + threadIdx.x;
    size_t total = (size_t)Kp * Np;
    if (idx >= total) return;
    int ntiles = Np >> 4;
    size_t tile = idx >> 9;
    int r = (int)(idx & 511);
    int lane = r >> 4, e = r & 15;
    int kt = (int)(tile / ntiles), nt = (int)(tile % ntiles);
    int n = nt * 16 + (lane & 15);
    int k = kt * 32 + ((lane >> 4) << 4) + e;
    float v = (k < Ko && n < No) ? W[(size_t)k * No + n] : 0.f;
    dst[idx] = f2bf(v);
}

// ---------- bf16 GEMM with WMMA + async-LDS B staging ----------
// Block = 4 waves; each wave owns one 16-row tile; all share 4 N-tiles (64 cols).
// Per K-step the block's B data is a contiguous 4KB slab of packed weights ->
// staged into an LDS ping-pong buffer with GLOBAL_LOAD_ASYNC_TO_LDS_B128.
template<bool RELU>
__global__ __launch_bounds__(128) void gemm_kernel(
    const unsigned short* __restrict__ A, int lda,
    const unsigned short* __restrict__ Bp,
    const float* __restrict__ bias, int Nb,
    unsigned short* __restrict__ C, int ldc,
    int Kp, int Np)
{
    __shared__ unsigned short Bs[2][4 * 512];   // 2 x 4KB ping-pong
    int lane = threadIdx.x;                     // 0..31 (wave32)
    int wave = threadIdx.y;                     // 0..3
    int tid  = wave * 32 + lane;                // 0..127
    int mt  = blockIdx.x * 4 + wave;            // 16-row tile
    int ntb = blockIdx.y * 4;                   // first 16-col tile
    int ntiles = Np >> 4;
    int mrow = lane & 15, hi = lane >> 4;
    int ktiles = Kp >> 5;

    const unsigned short* arow = A + (size_t)(mt * 16 + mrow) * lda;

    v8f zero = {0.f,0.f,0.f,0.f,0.f,0.f,0.f,0.f};
    v8f acc0 = zero, acc1 = zero, acc2 = zero, acc3 = zero;

    // ---- prologue: stage kt=0 ----
    {
        const unsigned short* src = Bp + ((size_t)0 * ntiles + ntb) * 512 + tid * 16;
        unsigned short* dst = &Bs[0][tid * 16];
        async_b128(src, dst);
        async_b128(src + 8, dst + 8);
    }
    ABfrag a0, a1;
    {
        const unsigned short* ap = arow + 16 * hi;
        a0.q[0] = *(const uint4*)(ap);
        a0.q[1] = *(const uint4*)(ap + 8);
    }
    wait_async0();
    __syncthreads();

    int kt = 0;
    while (true) {
        // ---- even step: compute (a0, Bs[0]); prefetch kt+1 into (a1, Bs[1]) ----
        if (kt + 1 < ktiles) {
            const unsigned short* src = Bp + ((size_t)(kt + 1) * ntiles + ntb) * 512 + tid * 16;
            unsigned short* dst = &Bs[1][tid * 16];
            async_b128(src, dst);
            async_b128(src + 8, dst + 8);
            const unsigned short* ap = arow + (kt + 1) * 32 + 16 * hi;
            a1.q[0] = *(const uint4*)(ap);
            a1.q[1] = *(const uint4*)(ap + 8);
        }
        {
            const unsigned short* bl = &Bs[0][lane * 16];
            ABfrag b0, b1, b2, b3;                     // load all fragments first:
            b0.q[0] = *(const uint4*)(bl);             // one dscnt wait, then 4
            b0.q[1] = *(const uint4*)(bl + 8);         // back-to-back WMMAs
            b1.q[0] = *(const uint4*)(bl + 512);
            b1.q[1] = *(const uint4*)(bl + 520);
            b2.q[0] = *(const uint4*)(bl + 1024);
            b2.q[1] = *(const uint4*)(bl + 1032);
            b3.q[0] = *(const uint4*)(bl + 1536);
            b3.q[1] = *(const uint4*)(bl + 1544);
            acc0 = __builtin_amdgcn_wmma_f32_16x16x32_bf16(false, a0.v, false, b0.v, (short)0, acc0, false, false);
            acc1 = __builtin_amdgcn_wmma_f32_16x16x32_bf16(false, a0.v, false, b1.v, (short)0, acc1, false, false);
            acc2 = __builtin_amdgcn_wmma_f32_16x16x32_bf16(false, a0.v, false, b2.v, (short)0, acc2, false, false);
            acc3 = __builtin_amdgcn_wmma_f32_16x16x32_bf16(false, a0.v, false, b3.v, (short)0, acc3, false, false);
        }
        wait_async0();
        __syncthreads();
        ++kt; if (kt >= ktiles) break;

        // ---- odd step: compute (a1, Bs[1]); prefetch kt+1 into (a0, Bs[0]) ----
        if (kt + 1 < ktiles) {
            const unsigned short* src = Bp + ((size_t)(kt + 1) * ntiles + ntb) * 512 + tid * 16;
            unsigned short* dst = &Bs[0][tid * 16];
            async_b128(src, dst);
            async_b128(src + 8, dst + 8);
            const unsigned short* ap = arow + (kt + 1) * 32 + 16 * hi;
            a0.q[0] = *(const uint4*)(ap);
            a0.q[1] = *(const uint4*)(ap + 8);
        }
        {
            const unsigned short* bl = &Bs[1][lane * 16];
            ABfrag b0, b1, b2, b3;
            b0.q[0] = *(const uint4*)(bl);
            b0.q[1] = *(const uint4*)(bl + 8);
            b1.q[0] = *(const uint4*)(bl + 512);
            b1.q[1] = *(const uint4*)(bl + 520);
            b2.q[0] = *(const uint4*)(bl + 1024);
            b2.q[1] = *(const uint4*)(bl + 1032);
            b3.q[0] = *(const uint4*)(bl + 1536);
            b3.q[1] = *(const uint4*)(bl + 1544);
            acc0 = __builtin_amdgcn_wmma_f32_16x16x32_bf16(false, a1.v, false, b0.v, (short)0, acc0, false, false);
            acc1 = __builtin_amdgcn_wmma_f32_16x16x32_bf16(false, a1.v, false, b1.v, (short)0, acc1, false, false);
            acc2 = __builtin_amdgcn_wmma_f32_16x16x32_bf16(false, a1.v, false, b2.v, (short)0, acc2, false, false);
            acc3 = __builtin_amdgcn_wmma_f32_16x16x32_bf16(false, a1.v, false, b3.v, (short)0, acc3, false, false);
        }
        wait_async0();
        __syncthreads();
        ++kt; if (kt >= ktiles) break;
    }

    // ---- epilogue: lane holds col = tile*16 + (lane&15); rows 8*hi + r ----
    v8f accs[4] = { acc0, acc1, acc2, acc3 };
#pragma unroll
    for (int j = 0; j < 4; ++j) {
        int n = (ntb + j) * 16 + mrow;
        float bn = (n < Nb) ? bias[n] : 0.f;
        unsigned short* cp = C + n;
#pragma unroll
        for (int r = 0; r < 8; ++r) {
            int mg = mt * 16 + hi * 8 + r;
            float v = accs[j][r] + bn;
            if (RELU) v = v > 0.f ? v : 0.f;
            cp[(size_t)mg * ldc] = f2bf(v);
        }
    }
}

// ---------- edge kernels ----------
// nb = 0.5*(x_i + x_j) into A1 cols[0:200); zero pad cols[400:416)
__global__ void ker_nb(const float* __restrict__ X, const int* __restrict__ ei,
                       const int* __restrict__ ej, unsigned short* __restrict__ A1) {
    int e = blockIdx.x, t = threadIdx.x;
    int i = ei[e], j = ej[e];
    if (t < 50) {
        float4 xi = *(const float4*)(X + (size_t)i * FF + 4 * t);
        float4 xj = *(const float4*)(X + (size_t)j * FF + 4 * t);
        ushort4 o = make_ushort4(f2bf(0.5f*(xi.x+xj.x)), f2bf(0.5f*(xi.y+xj.y)),
                                 f2bf(0.5f*(xi.z+xj.z)), f2bf(0.5f*(xi.w+xj.w)));
        *(ushort4*)(A1 + (size_t)e * 416 + 4 * t) = o;
    } else if (t < 54) {
        *(ushort4*)(A1 + (size_t)e * 416 + 400 + 4 * (t - 50)) = make_ushort4(0,0,0,0);
    }
}

// per-node segment max over nb (edges grouped by sorted edge_i)
__global__ void ker_segmax(const unsigned short* __restrict__ A1, const int* __restrict__ off,
                           unsigned short* __restrict__ G) {
    int n = blockIdx.x, t = threadIdx.x;
    if (t >= 50) return;
    int s = off[n], e1 = off[n + 1];
    float m0 = -3.4e38f, m1 = m0, m2 = m0, m3 = m0;
    for (int e = s; e < e1; ++e) {
        ushort4 v = *(const ushort4*)(A1 + (size_t)e * 416 + 4 * t);
        m0 = fmaxf(m0, bf2f(v.x)); m1 = fmaxf(m1, bf2f(v.y));
        m2 = fmaxf(m2, bf2f(v.z)); m3 = fmaxf(m3, bf2f(v.w));
    }
    if (s == e1) { m0 = m1 = m2 = m3 = 0.f; }   // empty segment: never gathered
    *(ushort4*)(G + (size_t)n * FF + 4 * t) =
        make_ushort4(f2bf(m0), f2bf(m1), f2bf(m2), f2bf(m3));
}

// gat = gmax[edge_i] into A1 cols[200:400)
__global__ void ker_gat(unsigned short* __restrict__ A1, const int* __restrict__ ei,
                        const unsigned short* __restrict__ G) {
    int e = blockIdx.x, t = threadIdx.x;
    if (t >= 50) return;
    int i = ei[e];
    ushort4 v = *(const ushort4*)(G + (size_t)i * FF + 4 * t);
    *(ushort4*)(A1 + (size_t)e * 416 + 200 + 4 * t) = v;
}

// A2 = [x_i | x_j | (Ef later) | 0 pad]
__global__ void ker_a2(const float* __restrict__ X, const int* __restrict__ ei,
                       const int* __restrict__ ej, unsigned short* __restrict__ A2) {
    int e = blockIdx.x, t = threadIdx.x;
    int i = ei[e], j = ej[e];
    if (t < 50) {
        float4 xi = *(const float4*)(X + (size_t)i * FF + 4 * t);
        float4 xj = *(const float4*)(X + (size_t)j * FF + 4 * t);
        *(ushort4*)(A2 + (size_t)e * 416 + 4 * t) =
            make_ushort4(f2bf(xi.x), f2bf(xi.y), f2bf(xi.z), f2bf(xi.w));
        *(ushort4*)(A2 + (size_t)e * 416 + 200 + 4 * t) =
            make_ushort4(f2bf(xj.x), f2bf(xj.y), f2bf(xj.z), f2bf(xj.w));
    } else if (t < 54) {
        *(ushort4*)(A2 + (size_t)e * 416 + 400 + 4 * (t - 50)) = make_ushort4(0,0,0,0);
    }
}

// Ef = H2 @ W2(128x1) + b2, written into A2 col 400 (overwrites pad)
__global__ void ker_ef(const unsigned short* __restrict__ H2, const float* __restrict__ W2,
                       const float* __restrict__ b2, unsigned short* __restrict__ A2) {
    int e = blockIdx.x * blockDim.x + threadIdx.x;
    if (e >= EE) return;
    const unsigned short* h = H2 + (size_t)e * 128;
    float acc = 0.f;
#pragma unroll 8
    for (int k = 0; k < 128; ++k) acc += bf2f(h[k]) * W2[k];
    A2[(size_t)e * 416 + 400] = f2bf(acc + b2[0]);
}

// Xout = X + segment_mean(M, edge_i);  M is E x 256 bf16 (cols 0..199 valid)
__global__ void ker_aggr(const unsigned short* __restrict__ M, const int* __restrict__ off,
                         const float* __restrict__ X, float* __restrict__ Xo) {
    int n = blockIdx.x, t = threadIdx.x;
    if (t >= 50) return;
    int s = off[n], e1 = off[n + 1];
    float a0 = 0, a1 = 0, a2 = 0, a3 = 0;
    for (int e = s; e < e1; ++e) {
        ushort4 v = *(const ushort4*)(M + (size_t)e * 256 + 4 * t);
        a0 += bf2f(v.x); a1 += bf2f(v.y); a2 += bf2f(v.z); a3 += bf2f(v.w);
    }
    float inv = 1.f / fmaxf((float)(e1 - s), 1.f);
    float4 x = *(const float4*)(X + (size_t)n * FF + 4 * t);
    float4 o = make_float4(x.x + a0 * inv, x.y + a1 * inv, x.z + a2 * inv, x.w + a3 * inv);
    *(float4*)(Xo + (size_t)n * FF + 4 * t) = o;
}

// per-graph mean pooling of node features
__global__ void ker_pool(const float* __restrict__ X, const int* __restrict__ boff,
                         float* __restrict__ gp) {
    int b = blockIdx.x, t = threadIdx.x;
    if (t >= 50) return;
    int s = boff[b], e1 = boff[b + 1];
    float a0 = 0, a1 = 0, a2 = 0, a3 = 0;
    for (int n = s; n < e1; ++n) {
        float4 v = *(const float4*)(X + (size_t)n * FF + 4 * t);
        a0 += v.x; a1 += v.y; a2 += v.z; a3 += v.w;
    }
    float inv = 1.f / fmaxf((float)(e1 - s), 1.f);
    *(float4*)(gp + (size_t)b * FF + 4 * t) = make_float4(a0*inv, a1*inv, a2*inv, a3*inv);
}

// conv1d over feature axis (kernel 3, SAME, 1->64 ch) + relu -> x3 (B x 12800 bf16)
__global__ void ker_conv1d(const float* __restrict__ gp, const float* __restrict__ cw,
                           const float* __restrict__ cb, unsigned short* __restrict__ x3) {
    int b = blockIdx.x;
    for (int idx = threadIdx.x; idx < 12800; idx += blockDim.x) {
        int w = idx >> 6, o = idx & 63;
        float acc = 0.f;
#pragma unroll
        for (int d = 0; d < 3; ++d) {
            int ww = w + d - 1;
            if (ww >= 0 && ww < FF) acc += gp[(size_t)b * FF + ww] * cw[d * 64 + o];
        }
        acc += cb[o];
        x3[(size_t)b * 12800 + idx] = f2bf(fmaxf(acc, 0.f));
    }
}

// ---------- small concat / fill kernels (head is tiny) ----------
__global__ void cat_f32(unsigned short* dst, int ldd, int col0,
                        const float* src, int lds_, int rows, int cols) {
    int idx = blockIdx.x * 256 + threadIdx.x;
    if (idx >= rows * cols) return;
    int r = idx / cols, c = idx % cols;
    dst[(size_t)r * ldd + col0 + c] = f2bf(src[(size_t)r * lds_ + c]);
}
__global__ void cat_u16(unsigned short* dst, int ldd, int col0,
                        const unsigned short* src, int lds_, int rows, int cols) {
    int idx = blockIdx.x * 256 + threadIdx.x;
    if (idx >= rows * cols) return;
    int r = idx / cols, c = idx % cols;
    dst[(size_t)r * ldd + col0 + c] = src[(size_t)r * lds_ + c];
}
__global__ void zero_u16(unsigned short* dst, int ldd, int col0, int rows, int cols) {
    int idx = blockIdx.x * 256 + threadIdx.x;
    if (idx >= rows * cols) return;
    int r = idx / cols, c = idx % cols;
    dst[(size_t)r * ldd + col0 + c] = 0;
}

// final 256 -> 1 layer, f32 output
__global__ void ker_d5(const unsigned short* __restrict__ h, const float* __restrict__ W,
                       const float* __restrict__ bb, float* __restrict__ out) {
    int b = blockIdx.x * blockDim.x + threadIdx.x;
    if (b >= BB) return;
    float acc = 0.f;
#pragma unroll 8
    for (int k = 0; k < 256; ++k) acc += bf2f(h[(size_t)b * 256 + k]) * W[k];
    out[b] = acc + bb[0];
}

// ---------- host helpers ----------
static void gemm(hipStream_t s, const unsigned short* A, int lda, const unsigned short* Bp,
                 const float* bias, int Nb, unsigned short* C, int ldc,
                 int M, int Kp, int Np, bool relu) {
    dim3 g(M / 64, Np / 64), b(32, 4);
    if (relu) gemm_kernel<true ><<<g, b, 0, s>>>(A, lda, Bp, bias, Nb, C, ldc, Kp, Np);
    else      gemm_kernel<false><<<g, b, 0, s>>>(A, lda, Bp, bias, Nb, C, ldc, Kp, Np);
}

static void pack(hipStream_t s, const float* W, int Ko, int No, int Kp, int Np,
                 unsigned short* dst) {
    size_t total = (size_t)Kp * Np;
    pack_w<<<(unsigned)((total + 255) / 256), 256, 0, s>>>(W, Ko, No, Kp, Np, dst);
}

struct ConvP {
    const float *eb0, *eb1, *eW2, *eb2, *sb0, *sb1, *sb2;
    const unsigned short *peW0, *peW1, *psW0, *psW1, *psW2;
};

static void run_conv(hipStream_t s, const float* Xin, float* Xout,
                     const int* ei, const int* ej, const int* noff,
                     unsigned short* A1, unsigned short* H1, unsigned short* H2,
                     unsigned short* G, const ConvP& p) {
    ker_nb    <<<EE, 64, 0, s>>>(Xin, ei, ej, A1);
    ker_segmax<<<NN, 64, 0, s>>>(A1, noff, G);
    ker_gat   <<<EE, 64, 0, s>>>(A1, ei, G);
    gemm(s, A1, 416, p.peW0, p.eb0, 256, H1, 256, EE, 416, 256, true);
    gemm(s, H1, 256, p.peW1, p.eb1, 128, H2, 128, EE, 256, 128, true);
    ker_a2<<<EE, 64, 0, s>>>(Xin, ei, ej, A1);          // A1 reused as snet input
    ker_ef<<<(EE + 255) / 256, 256, 0, s>>>(H2, p.eW2, p.eb2, A1);
    gemm(s, A1, 416, p.psW0, p.sb0, 256, H1, 256, EE, 416, 256, true);
    gemm(s, H1, 256, p.psW1, p.sb1, 128, H2, 128, EE, 256, 128, true);
    gemm(s, H2, 128, p.psW2, p.sb2, 200, A1, 256, EE, 128, 256, false); // M aliases A1
    ker_aggr<<<NN, 64, 0, s>>>(A1, noff, Xin, Xout);
}

extern "C" void kernel_launch(void* const* d_in, const int* in_sizes, int n_in,
                              void* d_out, int out_size, void* d_ws, size_t ws_size,
                              hipStream_t stream) {
    (void)in_sizes; (void)n_in; (void)out_size; (void)ws_size;
    // ---- inputs (setup_inputs() insertion order, params flattened recursively) ----
    const float* x       = (const float*)d_in[0];
    /* d_in[1] = e : unused by reference */
    const int*   edge_i  = (const int*)d_in[2];
    const int*   edge_j  = (const int*)d_in[3];
    const int*   batch_i = (const int*)d_in[4];
    int pi = 7;
    const float* c1eW0 = (const float*)d_in[pi++]; const float* c1eb0 = (const float*)d_in[pi++];
    const float* c1eW1 = (const float*)d_in[pi++]; const float* c1eb1 = (const float*)d_in[pi++];
    const float* c1eW2 = (const float*)d_in[pi++]; const float* c1eb2 = (const float*)d_in[pi++];
    const float* c1sW0 = (const float*)d_in[pi++]; const float* c1sb0 = (const float*)d_in[pi++];
    const float* c1sW1 = (const float*)d_in[pi++]; const float* c1sb1 = (const float*)d_in[pi++];
    const float* c1sW2 = (const float*)d_in[pi++]; const float* c1sb2 = (const float*)d_in[pi++];
    const float* c2eW0 = (const float*)d_in[pi++]; const float* c2eb0 = (const float*)d_in[pi++];
    const float* c2eW1 = (const float*)d_in[pi++]; const float* c2eb1 = (const float*)d_in[pi++];
    const float* c2eW2 = (const float*)d_in[pi++]; const float* c2eb2 = (const float*)d_in[pi++];
    const float* c2sW0 = (const float*)d_in[pi++]; const float* c2sb0 = (const float*)d_in[pi++];
    const float* c2sW1 = (const float*)d_in[pi++]; const float* c2sb1 = (const float*)d_in[pi++];
    const float* c2sW2 = (const float*)d_in[pi++]; const float* c2sb2 = (const float*)d_in[pi++];
    const float* convw = (const float*)d_in[pi++]; const float* convb = (const float*)d_in[pi++];
    const float* d1W = (const float*)d_in[pi++];   const float* d1b = (const float*)d_in[pi++];
    const float* d2W = (const float*)d_in[pi++];   const float* d2b = (const float*)d_in[pi++];
    const float* d3W = (const float*)d_in[pi++];   const float* d3b = (const float*)d_in[pi++];
    const float* d4W = (const float*)d_in[pi++];   const float* d4b = (const float*)d_in[pi++];
    const float* d5W = (const float*)d_in[pi++];   const float* d5b = (const float*)d_in[pi++];
    float* out = (float*)d_out;

    // ---- workspace carve ----
    uint8_t* wp = (uint8_t*)d_ws;
    auto take = [&](size_t bytes) -> void* {
        void* r = (void*)wp; wp += (bytes + 255) & ~(size_t)255; return r;
    };
    unsigned short* A1 = (unsigned short*)take((size_t)EE * 416 * 2);  // also snet-in & M
    unsigned short* H1 = (unsigned short*)take((size_t)EE * 256 * 2);
    unsigned short* H2 = (unsigned short*)take((size_t)EE * 128 * 2);
    unsigned short* G  = (unsigned short*)take((size_t)NN * 200 * 2);
    float* X1  = (float*)take((size_t)NN * FF * 4);
    float* X2  = (float*)take((size_t)NN * FF * 4);
    int*   noff = (int*)take((size_t)(NN + 1) * 4);
    int*   boff = (int*)take((size_t)(BB + 1) * 4);
    float* gp1 = (float*)take((size_t)BB * FF * 4);
    float* gp2 = (float*)take((size_t)BB * FF * 4);
    unsigned short* x3  = (unsigned short*)take((size_t)BB * 12800 * 2);
    unsigned short* d2A = (unsigned short*)take((size_t)BB * 736 * 2);
    unsigned short* d3A = (unsigned short*)take((size_t)BB * 1248 * 2);
    unsigned short* d4A = (unsigned short*)take((size_t)BB * 13824 * 2);
    unsigned short* h1h = (unsigned short*)take((size_t)BB * 512 * 2);
    unsigned short* h2h = (unsigned short*)take((size_t)BB * 1024 * 2);
    unsigned short* h3h = (unsigned short*)take((size_t)BB * 1024 * 2);
    unsigned short* h4h = (unsigned short*)take((size_t)BB * 256 * 2);
    unsigned short* p1eW0 = (unsigned short*)take((size_t)416 * 256 * 2);
    unsigned short* p1eW1 = (unsigned short*)take((size_t)256 * 128 * 2);
    unsigned short* p1sW0 = (unsigned short*)take((size_t)416 * 256 * 2);
    unsigned short* p1sW1 = (unsigned short*)take((size_t)256 * 128 * 2);
    unsigned short* p1sW2 = (unsigned short*)take((size_t)128 * 256 * 2);
    unsigned short* p2eW0 = (unsigned short*)take((size_t)416 * 256 * 2);
    unsigned short* p2eW1 = (unsigned short*)take((size_t)256 * 128 * 2);
    unsigned short* p2sW0 = (unsigned short*)take((size_t)416 * 256 * 2);
    unsigned short* p2sW1 = (unsigned short*)take((size_t)256 * 128 * 2);
    unsigned short* p2sW2 = (unsigned short*)take((size_t)128 * 256 * 2);
    unsigned short* pd1W  = (unsigned short*)take((size_t)12800 * 512 * 2);
    unsigned short* pd2W  = (unsigned short*)take((size_t)736 * 1024 * 2);
    unsigned short* pd3W  = (unsigned short*)take((size_t)1248 * 1024 * 2);
    unsigned short* pd4W  = (unsigned short*)take((size_t)13824 * 256 * 2);

    // ---- CSR offsets ----
    lb_kernel<<<(NN + 256) / 256, 256, 0, stream>>>(edge_i, EE, noff, NN);
    lb_kernel<<<(BB + 256) / 256, 256, 0, stream>>>(batch_i, NN, boff, BB);

    // ---- pack weights into WMMA tile layout ----
    pack(stream, c1eW0, 400, 256, 416, 256, p1eW0);
    pack(stream, c1eW1, 256, 128, 256, 128, p1eW1);
    pack(stream, c1sW0, 401, 256, 416, 256, p1sW0);
    pack(stream, c1sW1, 256, 128, 256, 128, p1sW1);
    pack(stream, c1sW2, 128, 200, 128, 256, p1sW2);
    pack(stream, c2eW0, 400, 256, 416, 256, p2eW0);
    pack(stream, c2eW1, 256, 128, 256, 128, p2eW1);
    pack(stream, c2sW0, 401, 256, 416, 256, p2sW0);
    pack(stream, c2sW1, 256, 128, 256, 128, p2sW1);
    pack(stream, c2sW2, 128, 200, 128, 256, p2sW2);
    pack(stream, d1W, 12800, 512, 12800, 512, pd1W);
    pack(stream, d2W,   712, 1024,  736, 1024, pd2W);
    pack(stream, d3W,  1224, 1024, 1248, 1024, pd3W);
    pack(stream, d4W, 13824,  256, 13824, 256, pd4W);

    // ---- crystal convs ----
    ConvP p1 = { c1eb0, c1eb1, c1eW2, c1eb2, c1sb0, c1sb1, c1sb2,
                 p1eW0, p1eW1, p1sW0, p1sW1, p1sW2 };
    ConvP p2 = { c2eb0, c2eb1, c2eW2, c2eb2, c2sb0, c2sb1, c2sb2,
                 p2eW0, p2eW1, p2sW0, p2sW1, p2sW2 };
    run_conv(stream, x,  X1, edge_i, edge_j, noff, A1, H1, H2, G, p1);
    run_conv(stream, X1, X2, edge_i, edge_j, noff, A1, H1, H2, G, p2);

    // ---- pooling ----
    ker_pool<<<BB, 64, 0, stream>>>(X1, boff, gp1);
    ker_pool<<<BB, 64, 0, stream>>>(X2, boff, gp2);

    // ---- head ----
    ker_conv1d<<<BB, 256, 0, stream>>>(gp2, convw, convb, x3);
    gemm(stream, x3, 12800, pd1W, d1b, 512, h1h, 512, BB, 12800, 512, true);

    cat_u16<<<(BB * 512 + 255) / 256, 256, 0, stream>>>(d2A, 736, 0, h1h, 512, BB, 512);
    cat_f32<<<(BB * 200 + 255) / 256, 256, 0, stream>>>(d2A, 736, 512, gp1, 200, BB, 200);
    zero_u16<<<(BB * 24 + 255) / 256, 256, 0, stream>>>(d2A, 736, 712, BB, 24);
    gemm(stream, d2A, 736, pd2W, d2b, 1024, h2h, 1024, BB, 736, 1024, true);

    cat_u16<<<(BB * 1024 + 255) / 256, 256, 0, stream>>>(d3A, 1248, 0, h2h, 1024, BB, 1024);
    cat_f32<<<(BB * 200 + 255) / 256, 256, 0, stream>>>(d3A, 1248, 1024, gp2, 200, BB, 200);
    zero_u16<<<(BB * 24 + 255) / 256, 256, 0, stream>>>(d3A, 1248, 1224, BB, 24);
    gemm(stream, d3A, 1248, pd3W, d3b, 1024, h3h, 1024, BB, 1248, 1024, true);

    cat_u16<<<(BB * 1024 + 255) / 256, 256, 0, stream>>>(d4A, 13824, 0, h3h, 1024, BB, 1024);
    cat_u16<<<(BB * 12800 + 255) / 256, 256, 0, stream>>>(d4A, 13824, 1024, x3, 12800, BB, 12800);
    gemm(stream, d4A, 13824, pd4W, d4b, 256, h4h, 256, BB, 13824, 256, true);

    ker_d5<<<1, 128, 0, stream>>>(h4h, d5W, d5b, out);
}
